// IterativeRetrievalReasoner_8555574854162
// MI455X (gfx1250) — matrix operand order repaired and checked by reading, not compiled, not run on hardware
//
#include <hip/hip_runtime.h>
#include <math.h>

// ---- problem constants (match reference) ----
constexpr int BB    = 256;     // batch
constexpr int DD    = 512;     // model dim
constexpr int QQ    = 128;     // query dim
constexpr int NN    = 500000;  // corpus size
constexpr int VV    = 50257;   // vocab
constexpr int HH    = 8;       // heads
constexpr int TOPK  = 8;
constexpr int HD    = 64;      // head dim
constexpr int STEPS = 5;

constexpr int SIMS_CN  = 512;                       // corpus rows per block
constexpr int SIMS_SUB = 64;                        // sims cols staged in LDS (64KB)
constexpr int NCHUNK   = (NN + SIMS_CN - 1) / SIMS_CN;  // 977

typedef __attribute__((ext_vector_type(16))) _Float16 v16h;
typedef __attribute__((ext_vector_type(8)))  float    v8f;

__device__ __forceinline__ float gelu_exact(float x) {
  return 0.5f * x * (1.0f + erff(x * 0.70710678118654752440f));
}

// =====================================================================
// Generic WMMA GEMM:  C[M,Nc] = act(A[M,Kd] @ W[Kd,Nc] + bias)
// One 16x16 output tile per wave; Kd multiple of 32; M multiple of 16.
// Branchless inner loop: column index clamped for loads, store predicated.
// act: 0 = none, 1 = exact GELU
// =====================================================================
__global__ void gemm_wmma_f16(const float* __restrict__ A,
                              const float* __restrict__ W,
                              const float* __restrict__ bias,
                              float* __restrict__ C,
                              int M, int Kd, int Nc, int act)
{
  const int lane = threadIdx.x & 31;
  const int wave = threadIdx.x >> 5;
  const int ntiles = (Nc + 15) >> 4;
  const int tile = blockIdx.x * 8 + wave;
  const int mt = tile / ntiles;
  const int nt = tile - mt * ntiles;
  if (mt * 16 >= M) return;                 // wave-uniform
  const int lr = lane & 15;
  const int hi = lane >> 4;
  const int row  = mt * 16 + lr;            // A row held by this lane
  const int col0 = nt * 16 + lr;            // logical C column
  const bool colok = (col0 < Nc);
  const int col = colok ? col0 : (Nc - 1);  // clamped: loads always in-bounds

  const float* __restrict__ ap = A + (unsigned)row * (unsigned)Kd;
  const float* wp = W + (unsigned)(hi << 4) * (unsigned)Nc + (unsigned)col;

  unsigned aoff[8], woff[8];
#pragma unroll
  for (int j = 0; j < 8; ++j) {
    // A fragment: contiguous k order (K = hi*16 + 2j), same as B
    aoff[j] = (unsigned)((hi << 4) + (j << 1));
    // B fragment: contiguous k order (K = hi*16 + 2j), row stride Nc
    woff[j] = (unsigned)(2 * j) * (unsigned)Nc;
  }

  v8f acc = {};
  for (int k0 = 0; k0 < Kd; k0 += 32) {
    v16h a, b;
#pragma unroll
    for (int j = 0; j < 8; ++j) {
      const float2 av = *(const float2*)(ap + (unsigned)k0 + aoff[j]);
      a[2 * j]     = (_Float16)av.x;
      a[2 * j + 1] = (_Float16)av.y;
      const float b0 = wp[woff[j]];
      const float b1 = wp[woff[j] + (unsigned)Nc];
      b[2 * j]     = (_Float16)b0;
      b[2 * j + 1] = (_Float16)b1;
    }
    wp += (unsigned)Nc * 32u;
    acc = __builtin_amdgcn_wmma_f32_16x16x32_f16(false, a, false, b,
                                                 (short)0, acc, false, false);
  }
  const float bv = (bias != nullptr) ? bias[col] : 0.0f;
  if (colok) {
#pragma unroll
    for (int r = 0; r < 8; ++r) {
      const int orow = mt * 16 + r + hi * 8;  // C/D layout: VGPR r -> M=r(+8)
      float x = acc[r] + bv;
      if (act == 1) x = gelu_exact(x);
      C[(unsigned)orow * (unsigned)Nc + (unsigned)col0] = x;
    }
  }
}

// =====================================================================
// Corpus key inverse norms: invn[n] = 1 / max(||keys[n,:]||, 1e-12)
// one wave per row (128 floats = 32 float4, one per lane)
// =====================================================================
__global__ void inv_norm_kernel(const float* __restrict__ keys,
                                float* __restrict__ invn, int Ntot)
{
  const int lane = threadIdx.x & 31;
  const int wave = threadIdx.x >> 5;
  const int n = blockIdx.x * 8 + wave;
  if (n >= Ntot) return;
  const float4 v = ((const float4*)(keys + (unsigned)n * QQ))[lane];
  float s = v.x * v.x + v.y * v.y + v.z * v.z + v.w * v.w;
#pragma unroll
  for (int o = 16; o > 0; o >>= 1) s += __shfl_xor(s, o, 32);
  if (lane == 0) invn[n] = 1.0f / fmaxf(sqrtf(s), 1e-12f);
}

// =====================================================================
// Fused sims + per-chunk top-8.
// Block: 256 threads (8 waves). Chunk: 512 corpus rows.
// S = query[256,128] @ keysChunkT, scaled by invn; staged 64 cols at a
// time in LDS; each thread keeps top-8 for its batch row.
// Loads are branchless (clamped row index, OOB columns scaled to 0 and
// excluded in the scan).
// =====================================================================
__global__ void sims_topk_kernel(const float* __restrict__ Qm,     // [256,128]
                                 const float* __restrict__ keys,   // [N,128]
                                 const float* __restrict__ invn,   // [N]
                                 float* __restrict__ pv,           // [NCHUNK,256,8]
                                 int*   __restrict__ pi,
                                 int Ntot)
{
  __shared__ float S[256 * SIMS_SUB];       // 64 KB
  const int t    = threadIdx.x;
  const int lane = t & 31;
  const int wave = t >> 5;
  const int lr   = lane & 15;
  const int hi   = lane >> 4;
  const int nt   = wave & 3;                // column tile within sub (4 tiles)
  const int m0   = (wave >> 2) * 8;         // row-tile start (waves split M)
  const int nbase = blockIdx.x * SIMS_CN;

  unsigned aoff[8], boff[8];
#pragma unroll
  for (int j = 0; j < 8; ++j) {
    aoff[j] = (unsigned)((hi << 4) + (j << 1));
    boff[j] = (unsigned)((hi << 4) + (j << 1));
  }

  float bestv[TOPK];
  int   besti[TOPK];
#pragma unroll
  for (int i = 0; i < TOPK; ++i) { bestv[i] = -INFINITY; besti[i] = 0; }

  for (int sub = 0; sub < SIMS_CN; sub += SIMS_SUB) {
    const int  colg0 = nbase + sub + nt * 16 + lr;  // corpus row (B column)
    const bool colok = (colg0 < Ntot);
    const int  colg  = colok ? colg0 : (Ntot - 1); // clamped
    const float sc   = colok ? invn[colg] : 0.0f;
    const int  pre   = (colg + SIMS_SUB < Ntot) ? (colg + SIMS_SUB) : colg;
    __builtin_prefetch(keys + (unsigned)pre * QQ, 0, 1);  // stream hint

    const float* __restrict__ kp = keys + (unsigned)colg * QQ;
    v16h bfrag[4];
#pragma unroll
    for (int ks = 0; ks < 4; ++ks) {
#pragma unroll
      for (int j = 0; j < 8; ++j) {
        const float2 kv = *(const float2*)(kp + (unsigned)(ks * 32) + boff[j]);
        bfrag[ks][2 * j]     = (_Float16)kv.x;
        bfrag[ks][2 * j + 1] = (_Float16)kv.y;
      }
    }
    for (int mt = m0; mt < m0 + 8; ++mt) {
      const float* __restrict__ qp = Qm + (unsigned)(mt * 16 + lr) * QQ;
      v8f acc = {};
#pragma unroll
      for (int ks = 0; ks < 4; ++ks) {
        v16h a;
#pragma unroll
        for (int j = 0; j < 8; ++j) {
          const float2 av = *(const float2*)(qp + (unsigned)(ks * 32) + aoff[j]);
          a[2 * j]     = (_Float16)av.x;
          a[2 * j + 1] = (_Float16)av.y;
        }
        acc = __builtin_amdgcn_wmma_f32_16x16x32_f16(false, a, false, bfrag[ks],
                                                     (short)0, acc, false, false);
      }
#pragma unroll
      for (int r = 0; r < 8; ++r)
        S[(mt * 16 + r + hi * 8) * SIMS_SUB + nt * 16 + lr] = acc[r] * sc;
    }
    __syncthreads();
    // per-thread (batch row t) scan of the staged 64 columns
    for (int c = 0; c < SIMS_SUB; ++c) {
      const int n = nbase + sub + c;
      const float vle = (n < Ntot) ? S[t * SIMS_SUB + c] : -INFINITY;
      if (vle > bestv[TOPK - 1]) {
        int p = TOPK - 1;
        while (p > 0 && vle > bestv[p - 1]) {
          bestv[p] = bestv[p - 1]; besti[p] = besti[p - 1]; --p;
        }
        bestv[p] = vle; besti[p] = n;
      }
    }
    __syncthreads();
  }
  const unsigned off = ((unsigned)blockIdx.x * 256u + (unsigned)t) * TOPK;
#pragma unroll
  for (int i = 0; i < TOPK; ++i) { pv[off + i] = bestv[i]; pi[off + i] = besti[i]; }
}

// =====================================================================
// Merge per-chunk top-8 lists to final top-8 per batch row.
// One block (256 threads) per batch row; tree-merge of sorted lists.
// =====================================================================
__global__ void topk_merge_kernel(const float* __restrict__ pv,
                                  const int*   __restrict__ pi,
                                  int nchunk, int* __restrict__ outi)
{
  __shared__ float sv[256 * TOPK];
  __shared__ int   si[256 * TOPK];
  const int b = blockIdx.x;
  const int t = threadIdx.x;
  float bvv[TOPK]; int bii[TOPK];
#pragma unroll
  for (int i = 0; i < TOPK; ++i) { bvv[i] = -INFINITY; bii[i] = 0; }
  const int total = nchunk * TOPK;
  for (int i = t; i < total; i += 256) {
    const int chunk = i >> 3, s = i & 7;
    const float v = pv[((unsigned)chunk * 256u + (unsigned)b) * TOPK + s];
    if (v > bvv[TOPK - 1]) {
      const int id = pi[((unsigned)chunk * 256u + (unsigned)b) * TOPK + s];
      int p = TOPK - 1;
      while (p > 0 && v > bvv[p - 1]) { bvv[p] = bvv[p - 1]; bii[p] = bii[p - 1]; --p; }
      bvv[p] = v; bii[p] = id;
    }
  }
#pragma unroll
  for (int i = 0; i < TOPK; ++i) { sv[t * TOPK + i] = bvv[i]; si[t * TOPK + i] = bii[i]; }
  __syncthreads();
  for (int stride = 128; stride > 0; stride >>= 1) {
    if (t < stride) {
      float a_[TOPK], c_[TOPK], m_[TOPK];
      int   ai_[TOPK], ci_[TOPK], mi_[TOPK];
#pragma unroll
      for (int i = 0; i < TOPK; ++i) {
        a_[i] = sv[t * TOPK + i];            ai_[i] = si[t * TOPK + i];
        c_[i] = sv[(t + stride) * TOPK + i]; ci_[i] = si[(t + stride) * TOPK + i];
      }
      int pa = 0, pb = 0;
#pragma unroll
      for (int i = 0; i < TOPK; ++i) {
        if (pb >= TOPK || (pa < TOPK && a_[pa] >= c_[pb])) { m_[i] = a_[pa]; mi_[i] = ai_[pa]; ++pa; }
        else                                               { m_[i] = c_[pb]; mi_[i] = ci_[pb]; ++pb; }
      }
#pragma unroll
      for (int i = 0; i < TOPK; ++i) { sv[t * TOPK + i] = m_[i]; si[t * TOPK + i] = mi_[i]; }
    }
    __syncthreads();
  }
  if (t == 0) {
#pragma unroll
    for (int i = 0; i < TOPK; ++i) outi[b * TOPK + i] = si[i];
  }
}

// =====================================================================
// Gather docs[b*K+k, :] = corpus_values[idx[b*K+k], :]  (2KB per block)
// CDNA5 async path: global -> LDS (async) -> global (async), 16B/lane.
// lbuf is the only LDS object in this kernel => its LDS offset is 0, so
// a lane's LDS byte address is simply t*16.
// =====================================================================
__global__ void gather_docs_kernel(const float* __restrict__ cv,
                                   const int* __restrict__ idx,
                                   float* __restrict__ docs)
{
  __shared__ __align__(16) float lbuf[DD];   // 2 KB staging
  const int bk = blockIdx.x;                 // 0..B*K-1
  const int t  = threadIdx.x;                // 0..127
  const int id = idx[bk];

  // opaque use pins the LDS allocation
  const float* lp = &lbuf[0];
  asm volatile("" :: "v"(lp) : "memory");

  const unsigned lds_off = (unsigned)t * 16u;
  const unsigned long long gsrc =
      (unsigned long long)(uintptr_t)(cv + (unsigned)id * DD + t * 4);
  const unsigned long long gdst =
      (unsigned long long)(uintptr_t)(docs + (unsigned)bk * DD + t * 4);

  asm volatile("global_load_async_to_lds_b128 %0, %1, off"
               :: "v"(lds_off), "v"(gsrc) : "memory");
  asm volatile("s_wait_asynccnt 0x0" ::: "memory");
  asm volatile("global_store_async_from_lds_b128 %0, %1, off"
               :: "v"(gdst), "v"(lds_off) : "memory");
  asm volatile("s_wait_asynccnt 0x0" ::: "memory");
}

// LayerNorm over rows of length L; one block per row (256 threads)
__global__ void layernorm_kernel(const float* __restrict__ x,
                                 const float* __restrict__ g,
                                 const float* __restrict__ bta,
                                 float* __restrict__ y, int L)
{
  __shared__ float rs[256];
  const int row = blockIdx.x, t = threadIdx.x;
  float s = 0.0f;
  for (int i = t; i < L; i += 256) s += x[(unsigned)row * L + i];
  rs[t] = s; __syncthreads();
  for (int st = 128; st > 0; st >>= 1) { if (t < st) rs[t] += rs[t + st]; __syncthreads(); }
  const float mean = rs[0] / (float)L; __syncthreads();
  float s2 = 0.0f;
  for (int i = t; i < L; i += 256) { const float d = x[(unsigned)row * L + i] - mean; s2 += d * d; }
  rs[t] = s2; __syncthreads();
  for (int st = 128; st > 0; st >>= 1) { if (t < st) rs[t] += rs[t + st]; __syncthreads(); }
  const float inv = rsqrtf(rs[0] / (float)L + 1e-5f);
  for (int i = t; i < L; i += 256)
    y[(unsigned)row * L + i] = (x[(unsigned)row * L + i] - mean) * inv * g[i] + bta[i];
}

// query = l2norm(base * (use_gate ? sigmoid(gatelin) : 1)); block per row, 128 thr
__global__ void gate_norm_kernel(const float* __restrict__ base,
                                 const float* __restrict__ gatelin,
                                 float* __restrict__ query, int use_gate)
{
  __shared__ float rs[QQ];
  const int b = blockIdx.x, t = threadIdx.x;
  float x = base[b * QQ + t];
  if (use_gate) x *= 1.0f / (1.0f + expf(-gatelin[b * QQ + t]));
  rs[t] = x * x; __syncthreads();
  for (int s = 64; s > 0; s >>= 1) { if (t < s) rs[t] += rs[t + s]; __syncthreads(); }
  const float nrm = sqrtf(rs[0]);
  query[b * QQ + t] = x / fmaxf(nrm, 1e-12f);
}

// out[row] = [a[row] ; c[row]]
__global__ void concat_kernel(const float* __restrict__ a,
                              const float* __restrict__ c,
                              float* __restrict__ out, int D1, int D2)
{
  const int row = blockIdx.x, t = threadIdx.x;
  for (int i = t; i < D1; i += 256) out[(unsigned)row * (D1 + D2) + i] = a[(unsigned)row * D1 + i];
  for (int i = t; i < D2; i += 256) out[(unsigned)row * (D1 + D2) + D1 + i] = c[(unsigned)row * D2 + i];
}

// cross-attention: 1 query token over K docs; block = 64 threads per (b,h)
__global__ void attn_kernel(const float* __restrict__ qb,   // [B, D]
                            const float* __restrict__ kb,   // [B*K, D]
                            const float* __restrict__ vb,   // [B*K, D]
                            float* __restrict__ out)        // [B, D]
{
  __shared__ float red[HD];
  __shared__ float lg[TOPK];
  const int b = blockIdx.x / HH, h = blockIdx.x % HH;
  const int t = threadIdx.x;                       // 0..63
  const float qv = qb[(unsigned)b * DD + h * HD + t];
  for (int j = 0; j < TOPK; ++j) {
    red[t] = qv * kb[(unsigned)(b * TOPK + j) * DD + h * HD + t];
    __syncthreads();
    for (int s = 32; s > 0; s >>= 1) { if (t < s) red[t] += red[t + s]; __syncthreads(); }
    if (t == 0) lg[j] = red[0] * 0.125f;           // 1/sqrt(64)
    __syncthreads();
  }
  float mx = -INFINITY;
#pragma unroll
  for (int j = 0; j < TOPK; ++j) mx = fmaxf(mx, lg[j]);
  float w[TOPK], se = 0.0f;
#pragma unroll
  for (int j = 0; j < TOPK; ++j) { w[j] = expf(lg[j] - mx); se += w[j]; }
  float o = 0.0f;
#pragma unroll
  for (int j = 0; j < TOPK; ++j)
    o += (w[j] / se) * vb[(unsigned)(b * TOPK + j) * DD + h * HD + t];
  out[(unsigned)b * DD + h * HD + t] = o;
}

// ctx = first ? thought : 0.7*thought + 0.3*ctx
__global__ void ctx_update_kernel(const float* __restrict__ thought,
                                  float* __restrict__ ctx, int first)
{
  const int i = blockIdx.x * 256 + threadIdx.x;    // B*D elements
  const float tv = thought[i];
  ctx[i] = first ? tv : (0.7f * tv + 0.3f * ctx[i]);
}

// =====================================================================
// Host-side orchestration
// =====================================================================
static inline void launch_gemm(const float* A, const float* W, const float* bias,
                               float* C, int M, int Kd, int Nc, int act,
                               hipStream_t s)
{
  const int ntiles = (Nc + 15) / 16;
  const int tiles  = (M / 16) * ntiles;
  const int blocks = (tiles + 7) / 8;
  gemm_wmma_f16<<<blocks, 256, 0, s>>>(A, W, bias, C, M, Kd, Nc, act);
}

extern "C" void kernel_launch(void* const* d_in, const int* in_sizes, int n_in,
                              void* d_out, int out_size, void* d_ws, size_t ws_size,
                              hipStream_t stream) {
  // inputs in setup_inputs() order
  const float* input_repr = (const float*)d_in[0];
  const float* qg_w1 = (const float*)d_in[1];  const float* qg_b1 = (const float*)d_in[2];
  const float* qg_w2 = (const float*)d_in[3];  const float* qg_b2 = (const float*)d_in[4];
  const float* qg_ln_g = (const float*)d_in[5];const float* qg_ln_b = (const float*)d_in[6];
  const float* gate_w = (const float*)d_in[7]; const float* gate_b = (const float*)d_in[8];
  const float* corpus_keys   = (const float*)d_in[9];
  const float* corpus_values = (const float*)d_in[10];
  const float* wq = (const float*)d_in[11]; const float* wk = (const float*)d_in[12];
  const float* wv = (const float*)d_in[13];
  const float* bq = (const float*)d_in[14]; const float* bk = (const float*)d_in[15];
  const float* bv = (const float*)d_in[16];
  const float* wo = (const float*)d_in[17]; const float* bo = (const float*)d_in[18];
  const float* tm_w1 = (const float*)d_in[19]; const float* tm_b1 = (const float*)d_in[20];
  const float* tm_ln_g = (const float*)d_in[21]; const float* tm_ln_b = (const float*)d_in[22];
  const float* tm_w2 = (const float*)d_in[23]; const float* tm_b2 = (const float*)d_in[24];
  const float* in_w = (const float*)d_in[25]; const float* in_b = (const float*)d_in[26];
  const float* in_ln_g = (const float*)d_in[27]; const float* in_ln_b = (const float*)d_in[28];
  const float* ans_w1 = (const float*)d_in[29]; const float* ans_b1 = (const float*)d_in[30];
  const float* ans_w2 = (const float*)d_in[31]; const float* ans_b2 = (const float*)d_in[32];
  (void)in_sizes; (void)n_in; (void)out_size; (void)ws_size;

  // workspace carve-out (float units; all 16B aligned)
  float* p = (float*)d_ws;
  auto alloc = [&](size_t nf) { float* r = p; p += nf; return r; };
  float* invn    = alloc((size_t)NN);
  float* curA    = alloc((size_t)BB * DD);
  float* curB    = alloc((size_t)BB * DD);
  float* ctx     = alloc((size_t)BB * DD);
  float* tmpD    = alloc((size_t)BB * DD);
  float* tmpD2   = alloc((size_t)BB * DD);
  float* baseLin = alloc((size_t)BB * QQ);
  float* baseV   = alloc((size_t)BB * QQ);
  float* gatelin = alloc((size_t)BB * QQ);
  float* query   = alloc((size_t)BB * QQ);
  float* catbuf  = alloc((size_t)BB * 2 * DD);
  float* qbuf    = alloc((size_t)BB * DD);
  float* kbuf    = alloc((size_t)BB * TOPK * DD);
  float* vbuf    = alloc((size_t)BB * TOPK * DD);
  float* docs    = alloc((size_t)BB * TOPK * DD);
  float* attnout = alloc((size_t)BB * DD);
  float* aobuf   = alloc((size_t)BB * DD);
  float* pvals   = alloc((size_t)NCHUNK * BB * TOPK);
  int*   pidx    = (int*)alloc((size_t)NCHUNK * BB * TOPK);
  int*   topidx  = (int*)alloc((size_t)BB * TOPK);
  float* logits  = (float*)d_out;

  // corpus key inverse norms (once per call)
  inv_norm_kernel<<<(NN + 7) / 8, 256, 0, stream>>>(corpus_keys, invn, NN);

  // input projection: current = LN(GELU(x @ in_w + in_b))
  launch_gemm(input_repr, in_w, in_b, tmpD, BB, DD, DD, 1, stream);
  layernorm_kernel<<<BB, 256, 0, stream>>>(tmpD, in_ln_g, in_ln_b, curA, DD);

  float* cur = curA;
  float* nxt = curB;
  for (int step = 0; step < STEPS; ++step) {
    // ---- query generation ----
    launch_gemm(cur, qg_w1, qg_b1, tmpD, BB, DD, DD, 1, stream);        // GELU
    launch_gemm(tmpD, qg_w2, qg_b2, baseLin, BB, DD, QQ, 0, stream);
    layernorm_kernel<<<BB, 256, 0, stream>>>(baseLin, qg_ln_g, qg_ln_b, baseV, QQ);
    if (step > 0) {
      concat_kernel<<<BB, 256, 0, stream>>>(cur, ctx, catbuf, DD, DD);
      launch_gemm(catbuf, gate_w, gate_b, gatelin, BB, 2 * DD, QQ, 0, stream);
    }
    gate_norm_kernel<<<BB, QQ, 0, stream>>>(baseV, gatelin, query, step > 0 ? 1 : 0);

    // ---- retrieval: WMMA sims + fused top-8, then merge + async gather ----
    sims_topk_kernel<<<NCHUNK, 256, 0, stream>>>(query, corpus_keys, invn,
                                                 pvals, pidx, NN);
    topk_merge_kernel<<<BB, 256, 0, stream>>>(pvals, pidx, NCHUNK, topidx);
    gather_docs_kernel<<<BB * TOPK, 128, 0, stream>>>(corpus_values, topidx, docs);

    // ---- cross attention ----
    launch_gemm(cur,  wq, bq, qbuf, BB, DD, DD, 0, stream);
    launch_gemm(docs, wk, bk, kbuf, BB * TOPK, DD, DD, 0, stream);
    launch_gemm(docs, wv, bv, vbuf, BB * TOPK, DD, DD, 0, stream);
    attn_kernel<<<BB * HH, HD, 0, stream>>>(qbuf, kbuf, vbuf, attnout);
    launch_gemm(attnout, wo, bo, aobuf, BB, DD, DD, 0, stream);

    // ---- thought MLP ----
    const float* prior = (step == 0) ? cur : ctx;
    concat_kernel<<<BB, 256, 0, stream>>>(aobuf, prior, catbuf, DD, DD);
    launch_gemm(catbuf, tm_w1, tm_b1, tmpD, BB, 2 * DD, DD, 1, stream);  // GELU
    layernorm_kernel<<<BB, 256, 0, stream>>>(tmpD, tm_ln_g, tm_ln_b, tmpD2, DD);
    launch_gemm(tmpD2, tm_w2, tm_b2, nxt, BB, DD, DD, 0, stream);

    ctx_update_kernel<<<(BB * DD) / 256, 256, 0, stream>>>(nxt, ctx, step == 0 ? 1 : 0);
    float* t2 = cur; cur = nxt; nxt = t2;
  }

  // ---- answer head ----
  launch_gemm(cur, ans_w1, ans_b1, tmpD, BB, DD, DD, 1, stream);         // GELU
  launch_gemm(tmpD, ans_w2, ans_b2, logits, BB, DD, VV, 0, stream);
}
